// MultiHeadCrossAttentionLayer_8486855377194
// MI455X (gfx1250) — matrix operand order repaired and checked by reading, not compiled, not run on hardware
//
#include <hip/hip_runtime.h>

// MI455X / gfx1250, wave32. WMMA bf16 16x16x32, f32 accumulate.

typedef __attribute__((ext_vector_type(16))) __bf16 v16bf;
typedef __attribute__((ext_vector_type(8)))  float  v8f;

#define DEVINL __device__ __forceinline__

// ---------------- fragment loaders (ISA 7.12.2 layouts) ----------------

// A-matrix 16x32 bf16, row-major source, ld in elements.
// lane 0-15: row = lane,  K = kb(0)+[0..7] and 16+[0..7]
// lane16-31: row = lane-16, K = 8+[0..7] and 24+[0..7]
DEVINL v16bf load_frag_a(const __bf16* p, int ld, int lane) {
    const int r  = lane & 15;
    const int kb = (lane >> 4) << 3;            // 0 or 8
    const __bf16* q0 = p + (size_t)r * ld + kb; // 16B aligned (ld % 8 == 0)
    union { uint4 u[2]; v16bf v; } f;
    f.u[0] = *reinterpret_cast<const uint4*>(q0);
    f.u[1] = *reinterpret_cast<const uint4*>(q0 + 16);
    return f.v;
}

// A-fragment from f32 source with on-the-fly bf16 convert.
DEVINL v16bf load_frag_a_f32(const float* p, int ld, int lane) {
    const int r  = lane & 15;
    const int kb = (lane >> 4) << 3;
    const float* q0 = p + (size_t)r * ld + kb;
    float4 f0 = *reinterpret_cast<const float4*>(q0);
    float4 f1 = *reinterpret_cast<const float4*>(q0 + 4);
    float4 f2 = *reinterpret_cast<const float4*>(q0 + 16);
    float4 f3 = *reinterpret_cast<const float4*>(q0 + 20);
    v16bf v;
    v[0]=(__bf16)f0.x; v[1]=(__bf16)f0.y; v[2]=(__bf16)f0.z; v[3]=(__bf16)f0.w;
    v[4]=(__bf16)f1.x; v[5]=(__bf16)f1.y; v[6]=(__bf16)f1.z; v[7]=(__bf16)f1.w;
    v[8]=(__bf16)f2.x; v[9]=(__bf16)f2.y; v[10]=(__bf16)f2.z; v[11]=(__bf16)f2.w;
    v[12]=(__bf16)f3.x; v[13]=(__bf16)f3.y; v[14]=(__bf16)f3.z; v[15]=(__bf16)f3.w;
    return v;
}

// B-matrix 32x16: source is B^T row-major (row = N, col = K).
// lanes 0-15: N = lane, K = 0..7 and 16..23 ; lanes 16-31: N = lane-16, K = 8..15 and 24..31
DEVINL v16bf load_frag_b(const __bf16* p, int ld, int lane) {
    const int r  = lane & 15;
    const int kb = (lane >> 4) << 3;            // 0 or 8
    const __bf16* q0 = p + (size_t)r * ld + kb;
    union { uint4 u[2]; v16bf v; } f;
    f.u[0] = *reinterpret_cast<const uint4*>(q0);
    f.u[1] = *reinterpret_cast<const uint4*>(q0 + 16);
    return f.v;
}

DEVINL v8f wmma_bf16(v16bf a, v16bf b, v8f c) {
    return __builtin_amdgcn_wmma_f32_16x16x32_bf16(false, a, false, b,
                                                   (short)0, c, false, false);
}

// C/D layout: VGPR v -> lanes 0-15: (M=v, N=lane); lanes 16-31: (M=8+v, N=lane-16)

// ---------------- helper kernels ----------------

// dst[N,K] bf16  <- src[K,N] f32   (weight transpose + convert)
__global__ void mha_wtr_kernel(const float* __restrict__ src, __bf16* __restrict__ dst,
                               int K, int N) {
    int tid = blockIdx.x * 256 + threadIdx.x;
    if (tid >= K * N) return;
    int n = tid / K;
    int k = tid - n * K;
    dst[tid] = (__bf16)src[(size_t)k * N + n];
}

// VT[bh][dd][q] <- V[bh][q][dd]   (per-head value transpose, bf16)
__global__ void mha_vtr_kernel(const __bf16* __restrict__ V, __bf16* __restrict__ VT) {
    int tid = blockIdx.x * 256 + threadIdx.x;      // 128*65536 threads
    int bh  = tid >> 16;
    int idx = tid & 65535;
    int dd  = idx >> 9;
    int q   = idx & 511;
    VT[(size_t)bh * 65536 + (size_t)dd * 512 + q] =
        V[(size_t)bh * 65536 + (size_t)q * 128 + dd];
}

// ---------------- GEMM kernels (8 waves, 128x128 WG tile, 32x64 wave tile) ----

// Y[8192,1024] bf16 = X[8192,128] f32 @ W + bias   (WT = W^T bf16 [1024,128])
__global__ void mha_proj_kernel(const float* __restrict__ X, const __bf16* __restrict__ WT,
                                const float* __restrict__ bias, __bf16* __restrict__ Y) {
    const int lane = threadIdx.x & 31, wave = threadIdx.x >> 5;
    const int wm = wave & 3, wn = wave >> 2;
    const int m0 = blockIdx.x * 128 + wm * 32;
    const int n0 = blockIdx.y * 128 + wn * 64;
    v8f acc[2][4] = {};
    for (int k = 0; k < 128; k += 32) {
        v16bf a[2], b[4];
#pragma unroll
        for (int mi = 0; mi < 2; ++mi)
            a[mi] = load_frag_a_f32(X + (size_t)(m0 + 16 * mi) * 128 + k, 128, lane);
#pragma unroll
        for (int ni = 0; ni < 4; ++ni)
            b[ni] = load_frag_b(WT + (size_t)(n0 + 16 * ni) * 128 + k, 128, lane);
#pragma unroll
        for (int mi = 0; mi < 2; ++mi)
#pragma unroll
            for (int ni = 0; ni < 4; ++ni)
                acc[mi][ni] = wmma_bf16(a[mi], b[ni], acc[mi][ni]);
    }
    const int cn = lane & 15, cm = (lane >> 4) << 3;
#pragma unroll
    for (int mi = 0; mi < 2; ++mi)
#pragma unroll
        for (int ni = 0; ni < 4; ++ni) {
            int gm = m0 + mi * 16 + cm, gn = n0 + ni * 16 + cn;
            float bv = bias[gn];
#pragma unroll
            for (int v = 0; v < 8; ++v)
                Y[(size_t)(gm + v) * 1024 + gn] = (__bf16)(acc[mi][ni][v] + bv);
        }
}

// ST[bh][k][q] = scale * (Q[bh] @ K[bh]^T)[q][k]   (stored transposed, f32)
// Transposed store is vectorized: each lane's 8 accumulator values are
// contiguous along q, so emit two b128 stores per (mi,ni).
__global__ void mha_qk_kernel(const __bf16* __restrict__ Q, const __bf16* __restrict__ K,
                              float* __restrict__ ST, float scale) {
    const int bh = blockIdx.z;
    const __bf16* Qc = Q + (size_t)bh * 65536;
    const __bf16* Kc = K + (size_t)bh * 65536;
    float* Sc = ST + (size_t)bh * 262144;
    const int lane = threadIdx.x & 31, wave = threadIdx.x >> 5;
    const int wm = wave & 3, wn = wave >> 2;
    const int m0 = blockIdx.x * 128 + wm * 32;   // q
    const int n0 = blockIdx.y * 128 + wn * 64;   // key
    v8f acc[2][4] = {};
    for (int k = 0; k < 128; k += 32) {
        v16bf a[2], b[4];
#pragma unroll
        for (int mi = 0; mi < 2; ++mi)
            a[mi] = load_frag_a(Qc + (size_t)(m0 + 16 * mi) * 128 + k, 128, lane);
#pragma unroll
        for (int ni = 0; ni < 4; ++ni)
            b[ni] = load_frag_b(Kc + (size_t)(n0 + 16 * ni) * 128 + k, 128, lane);
#pragma unroll
        for (int mi = 0; mi < 2; ++mi)
#pragma unroll
            for (int ni = 0; ni < 4; ++ni)
                acc[mi][ni] = wmma_bf16(a[mi], b[ni], acc[mi][ni]);
    }
    const int cn = lane & 15, cm = (lane >> 4) << 3;
#pragma unroll
    for (int mi = 0; mi < 2; ++mi)
#pragma unroll
        for (int ni = 0; ni < 4; ++ni) {
            int gm = m0 + mi * 16 + cm, gn = n0 + ni * 16 + cn;
            float4 lo, hi;
            lo.x = acc[mi][ni][0] * scale; lo.y = acc[mi][ni][1] * scale;
            lo.z = acc[mi][ni][2] * scale; lo.w = acc[mi][ni][3] * scale;
            hi.x = acc[mi][ni][4] * scale; hi.y = acc[mi][ni][5] * scale;
            hi.z = acc[mi][ni][6] * scale; hi.w = acc[mi][ni][7] * scale;
            float* dst = Sc + (size_t)gn * 512 + gm;   // 32B aligned (gm % 8 == 0)
            *reinterpret_cast<float4*>(dst)     = lo;
            *reinterpret_cast<float4*>(dst + 4) = hi;
        }
}

// Row softmax over k per q; emits normalized exp as bf16 (still [k][q] layout).
__global__ void mha_softmax_kernel(const float* __restrict__ ST, __bf16* __restrict__ ET) {
    int tid = blockIdx.x * 256 + threadIdx.x;   // 128*512 threads
    int bh = tid >> 9, q = tid & 511;
    const float* S = ST + (size_t)bh * 262144 + q;
    __bf16* E = ET + (size_t)bh * 262144 + q;
    float m = -3.4e38f;
    for (int k = 0; k < 512; ++k) m = fmaxf(m, S[(size_t)k * 512]);
    float s = 0.f;
    for (int k = 0; k < 512; ++k) s += __expf(S[(size_t)k * 512] - m);
    float inv = 1.0f / s;
    for (int k = 0; k < 512; ++k)
        E[(size_t)k * 512] = (__bf16)(__expf(S[(size_t)k * 512] - m) * inv);
}

// AO[bh][k][dd] = (attn^T @ V)  = ET[bh][k][:] dot V[bh][:][dd], via VT.
__global__ void mha_pv_kernel(const __bf16* __restrict__ ET, const __bf16* __restrict__ VT,
                              __bf16* __restrict__ AO) {
    const int bh = blockIdx.z;
    const __bf16* Ec = ET + (size_t)bh * 262144;  // [k=512][q=512]
    const __bf16* Vc = VT + (size_t)bh * 65536;   // [dd=128][q=512]
    __bf16* Ac = AO + (size_t)bh * 65536;         // [k=512][dd=128]
    const int lane = threadIdx.x & 31, wave = threadIdx.x >> 5;
    const int wm = wave & 3, wn = wave >> 2;
    const int m0 = blockIdx.x * 128 + wm * 32;   // k index
    const int n0 = wn * 64;                      // dd index (N=128 total)
    v8f acc[2][4] = {};
    for (int k = 0; k < 512; k += 32) {
        v16bf a[2], b[4];
#pragma unroll
        for (int mi = 0; mi < 2; ++mi)
            a[mi] = load_frag_a(Ec + (size_t)(m0 + 16 * mi) * 512 + k, 512, lane);
#pragma unroll
        for (int ni = 0; ni < 4; ++ni)
            b[ni] = load_frag_b(Vc + (size_t)(n0 + 16 * ni) * 512 + k, 512, lane);
#pragma unroll
        for (int mi = 0; mi < 2; ++mi)
#pragma unroll
            for (int ni = 0; ni < 4; ++ni)
                acc[mi][ni] = wmma_bf16(a[mi], b[ni], acc[mi][ni]);
    }
    const int cn = lane & 15, cm = (lane >> 4) << 3;
#pragma unroll
    for (int mi = 0; mi < 2; ++mi)
#pragma unroll
        for (int ni = 0; ni < 4; ++ni) {
            int gm = m0 + mi * 16 + cm, gn = n0 + ni * 16 + cn;
#pragma unroll
            for (int v = 0; v < 8; ++v)
                Ac[(size_t)(gm + v) * 128 + gn] = (__bf16)acc[mi][ni][v];
        }
}

// OUT[8192,1024] f32 = AO[8192,1024] bf16 @ Wo + bo    (WoT = Wo^T bf16)
__global__ void mha_out_kernel(const __bf16* __restrict__ AO, const __bf16* __restrict__ WoT,
                               const float* __restrict__ bias, float* __restrict__ OUT) {
    const int lane = threadIdx.x & 31, wave = threadIdx.x >> 5;
    const int wm = wave & 3, wn = wave >> 2;
    const int m0 = blockIdx.x * 128 + wm * 32;
    const int n0 = blockIdx.y * 128 + wn * 64;
    v8f acc[2][4] = {};
    for (int k = 0; k < 1024; k += 32) {
        if (k + 32 < 1024) {
            __builtin_prefetch(AO + (size_t)m0 * 1024 + k + 32, 0, 1);
            __builtin_prefetch(WoT + (size_t)n0 * 1024 + k + 32, 0, 1);
        }
        v16bf a[2], b[4];
#pragma unroll
        for (int mi = 0; mi < 2; ++mi)
            a[mi] = load_frag_a(AO + (size_t)(m0 + 16 * mi) * 1024 + k, 1024, lane);
#pragma unroll
        for (int ni = 0; ni < 4; ++ni)
            b[ni] = load_frag_b(WoT + (size_t)(n0 + 16 * ni) * 1024 + k, 1024, lane);
#pragma unroll
        for (int mi = 0; mi < 2; ++mi)
#pragma unroll
            for (int ni = 0; ni < 4; ++ni)
                acc[mi][ni] = wmma_bf16(a[mi], b[ni], acc[mi][ni]);
    }
    const int cn = lane & 15, cm = (lane >> 4) << 3;
#pragma unroll
    for (int mi = 0; mi < 2; ++mi)
#pragma unroll
        for (int ni = 0; ni < 4; ++ni) {
            int gm = m0 + mi * 16 + cm, gn = n0 + ni * 16 + cn;
            float bv = bias[gn];
#pragma unroll
            for (int v = 0; v < 8; ++v)
                OUT[(size_t)(gm + v) * 1024 + gn] = acc[mi][ni][v] + bv;
        }
}

// ---------------- host ----------------

extern "C" void kernel_launch(void* const* d_in, const int* in_sizes, int n_in,
                              void* d_out, int out_size, void* d_ws, size_t ws_size,
                              hipStream_t stream) {
    const float* in1 = (const float*)d_in[0];
    const float* in2 = (const float*)d_in[1];
    const float* Wq  = (const float*)d_in[2];
    const float* bq  = (const float*)d_in[3];
    const float* Wk  = (const float*)d_in[4];
    const float* bk  = (const float*)d_in[5];
    const float* Wv  = (const float*)d_in[6];
    const float* bv  = (const float*)d_in[7];
    const float* Wo  = (const float*)d_in[8];
    const float* bo  = (const float*)d_in[9];

    char* w = (char*)d_ws;
    __bf16* WqT = (__bf16*)w; w += (size_t)131072 * 2;
    __bf16* WkT = (__bf16*)w; w += (size_t)131072 * 2;
    __bf16* WvT = (__bf16*)w; w += (size_t)131072 * 2;
    __bf16* WoT = (__bf16*)w; w += (size_t)1048576 * 2;
    __bf16* Qb  = (__bf16*)w; w += (size_t)8388608 * 2;
    __bf16* Kb  = (__bf16*)w; w += (size_t)8388608 * 2;
    __bf16* Vb  = (__bf16*)w; w += (size_t)8388608 * 2;
    __bf16* VT  = (__bf16*)w; w += (size_t)8388608 * 2;
    __bf16* AO  = (__bf16*)w; w += (size_t)8388608 * 2;
    __bf16* ET  = (__bf16*)w; w += (size_t)33554432 * 2;
    float*  ST  = (float*)w;  w += (size_t)33554432 * 4;   // total ~288 MB

    mha_wtr_kernel<<<512, 256, 0, stream>>>(Wq, WqT, 128, 1024);
    mha_wtr_kernel<<<512, 256, 0, stream>>>(Wk, WkT, 128, 1024);
    mha_wtr_kernel<<<512, 256, 0, stream>>>(Wv, WvT, 128, 1024);
    mha_wtr_kernel<<<4096, 256, 0, stream>>>(Wo, WoT, 1024, 1024);

    const float scale = 0.08838834764831845f;  // 1/sqrt(128)

    for (int half = 0; half < 2; ++half) {
        const float* xq = half ? in2 : in1;
        const float* xk = half ? in1 : in2;
        // projections (xv == xq in the reference)
        mha_proj_kernel<<<dim3(64, 8), 256, 0, stream>>>(xq, WqT, bq, Qb);
        mha_proj_kernel<<<dim3(64, 8), 256, 0, stream>>>(xk, WkT, bk, Kb);
        mha_proj_kernel<<<dim3(64, 8), 256, 0, stream>>>(xq, WvT, bv, Vb);
        mha_vtr_kernel<<<32768, 256, 0, stream>>>(Vb, VT);
        mha_qk_kernel<<<dim3(4, 4, 128), 256, 0, stream>>>(Qb, Kb, ST, scale);
        mha_softmax_kernel<<<256, 256, 0, stream>>>(ST, ET);
        mha_pv_kernel<<<dim3(4, 1, 128), 256, 0, stream>>>(ET, VT, AO);
        mha_out_kernel<<<dim3(64, 8), 256, 0, stream>>>(
            AO, WoT, bo, (float*)d_out + (size_t)half * 8388608);
    }
}
